// SelfAttention_33182917329333
// MI455X (gfx1250) — matrix operand order repaired and checked by reading, not compiled, not run on hardware
//
#include <hip/hip_runtime.h>

// ---------------------------------------------------------------------------
// MI455X (gfx1250) self-attention pipeline, all-bf16 GEMM operands:
//   convert:  query/W_in/W_out f32 -> bf16 (ws)
//   GEMM1:    q    = perm(query) x W_in^T        (wmma bf16, epi -> bf16)
//   GEMM2:    S    = q x q^T + causal mask       (epi -> f32 weights region)
//   softmax:  in place, row0 zeroed, + bf16 copy
//   GEMM3:    mix  = W x q   (via q^T, NT form)  (epi -> bf16)
//   GEMM4:    out  = selu([mix|q] x W_out^T)     (epi -> f32, [n,b,d] scatter)
// Tile staging uses the CDNA5 data movers: A tile via TENSOR_LOAD_TO_LDS
// (TDM descriptor with pad: +4 dwords per 16, matching the [64][20] LDS
// layout; s_wait_tensorcnt), B tile via per-lane global_load_async_to_lds_b128
// (s_wait_asynccnt). Compute: v_wmma_f32_16x16x32_bf16, f32 accumulation.
// ---------------------------------------------------------------------------

typedef __attribute__((ext_vector_type(16))) __bf16       v16bf;
typedef __attribute__((ext_vector_type(8)))  float        v8f;
typedef __attribute__((ext_vector_type(4)))  unsigned int u32x4;
typedef __attribute__((ext_vector_type(8)))  unsigned int u32x8;
typedef unsigned short u16;

union FragBF { v16bf v; unsigned int u[8]; };

// Two f32 -> two RNE bf16 packed in one dword (lo = a, hi = b).
__device__ __forceinline__ unsigned int pack_bf16(float a, float b) {
    unsigned int ua = __float_as_uint(a);
    unsigned int ub = __float_as_uint(b);
    ua += 0x7fffu + ((ua >> 16) & 1u);
    ub += 0x7fffu + ((ub >> 16) & 1u);
    return (ua >> 16) | (ub & 0xffff0000u);
}
__device__ __forceinline__ u16 bf16_of(float a) { return (u16)pack_bf16(a, 0.f); }

// ---------------------------------------------------------------------------
// TDM: load a 64-row x 32-element bf16 tile (row stride ld elements) into LDS
// at byte offset lds_off, padding 4 dwords after every 16 (row stride 80B).
// D# group0/group1 per ISA 8.3/8.4; 2D tile -> groups 2/3 disabled.
// ---------------------------------------------------------------------------
__device__ __forceinline__ void tdm_load_tile64x32(const u16* gsrc,
                                                   unsigned lds_off,
                                                   unsigned ld_elems) {
    unsigned long long ga = (unsigned long long)(__SIZE_TYPE__)gsrc;
    u32x4 g0;
    g0[0] = 1u;                                            // count=1, user mode
    g0[1] = lds_off;                                       // lds_addr
    g0[2] = (unsigned)ga;                                  // global_addr[31:0]
    g0[3] = (unsigned)((ga >> 32) & 0x1FFFFFFu) | (2u << 30); // ga[56:32] | type=2
    u32x8 g1;
    g1[0] = (1u << 16)        // data_size = 2B
          | (1u << 20)        // pad_enable
          | (3u << 22)        // pad_interval: 16 dwords
          | (3u << 25);       // pad_amount: 4 dwords
    g1[1] = (ld_elems & 0xFFFFu) << 16;                    // tensor_dim0 lo16
    g1[2] = (ld_elems >> 16);                              // dim0 hi | dim1 lo16 (=0)
    g1[3] = 0x10u | (32u << 16);                           // dim1 hi (0x100000) | tile_dim0=32
    g1[4] = 64u;                                           // tile_dim1=64, tile_dim2=0
    g1[5] = ld_elems;                                      // tensor_dim0_stride lo32
    g1[6] = 0u;                                            // stride hi | dim1_stride lo
    g1[7] = 0u;
    asm volatile("tensor_load_to_lds %0, %1" :: "s"(g0), "s"(g1) : "memory");
}

// Per-lane async copy: 16 bytes global -> LDS (ASYNCcnt tracked).
__device__ __forceinline__ void async_copy_b128(unsigned lds_off, const void* g) {
    unsigned long long ga = (unsigned long long)(__SIZE_TYPE__)g;
    asm volatile("global_load_async_to_lds_b128 %0, %1, off"
                 :: "v"(lds_off), "v"(ga) : "memory");
}
__device__ __forceinline__ void wait_asynccnt0() {
    asm volatile("s_wait_asynccnt 0" ::: "memory");
}

// ---------------------------------------------------------------------------
// One K=32 WMMA step: wave `warp` computes its 16x64 strip (4 tiles).
// LDS rows: 16 data dwords (32 bf16) + 4 pad dwords (stride 20).
// A frag (ISA 7.12.2 16-bit A 16x32): lane L -> row L&15,
//   dword i -> K-pair (i&3) + (i>>2)*8 + (L>>4)*4.
// B frag (32x16): lane L -> col L&15, dword i -> K-pair (i&3) + (i>>2)*8 + (L>>4)*4.
// ---------------------------------------------------------------------------
__device__ __forceinline__ void wmma_step(const unsigned int (*sA)[20],
                                          const unsigned int (*sB)[20],
                                          int warp, int lane, v8f* acc) {
    const int half = lane >> 4;
    const int r = (warp << 4) + (lane & 15);
    FragBF a;
#pragma unroll
    for (int i = 0; i < 8; ++i)
        a.u[i] = sA[r][(i & 3) + ((i >> 2) << 3) + (half << 2)];
#pragma unroll
    for (int t = 0; t < 4; ++t) {
        FragBF b;
        const int c = (t << 4) + (lane & 15);
#pragma unroll
        for (int i = 0; i < 8; ++i) b.u[i] = sB[c][(i & 3) + ((i >> 2) << 3) + (half << 2)];
        acc[t] = __builtin_amdgcn_wmma_f32_16x16x32_bf16(
            false, a.v, false, b.v, (short)0, acc[t], false, false);
    }
}

// Unified NT-form bf16 GEMM inner loop: C_tile += A[m0.., k] * B[n0.., k]^T.
__device__ __forceinline__ void gemm_nt_bf16(const u16* __restrict__ A, int lda,
                                             const u16* __restrict__ B, int ldb,
                                             int m0, int n0, int K,
                                             int tid, int lane, int warp,
                                             unsigned int (*sA)[20],
                                             unsigned int (*sB)[20], v8f* acc) {
    const unsigned ldsA = (unsigned)(__SIZE_TYPE__)&sA[0][0];
    const unsigned ldsB = (unsigned)(__SIZE_TYPE__)&sB[0][0];
    for (int k0 = 0; k0 < K; k0 += 32) {
        if (warp == 0)                                        // A tile: one TDM DMA
            tdm_load_tile64x32(A + (size_t)m0 * lda + k0, ldsA, (unsigned)lda);
#pragma unroll
        for (int i = 0; i < 2; ++i) {                         // B tile: per-lane async
            const int idx = tid + i * 128;                    // 256 x 16B = 4KB
            const int row = idx >> 2, quad = idx & 3;
            async_copy_b128(ldsB + (unsigned)(row * 80 + quad * 16),
                            B + (size_t)(n0 + row) * ldb + k0 + quad * 8);
        }
        if (warp == 0) __builtin_amdgcn_s_wait_tensorcnt(0);
        wait_asynccnt0();
        __syncthreads();
        wmma_step(sA, sB, warp, lane, acc);
        __syncthreads();
    }
}

#define ZERO_ACC(acc) \
    v8f acc[4];       \
    { v8f z = {0.f,0.f,0.f,0.f,0.f,0.f,0.f,0.f}; acc[0]=z; acc[1]=z; acc[2]=z; acc[3]=z; }

// --------------------------- conversion kernels ----------------------------
__global__ void __launch_bounds__(256)
k_pack(const float* __restrict__ src, unsigned int* __restrict__ dst, int npairs) {
    int i = blockIdx.x * 256 + threadIdx.x;
    if (i < npairs) dst[i] = pack_bf16(src[2 * i], src[2 * i + 1]);
}

// qin[r=b*2048+n, d] = query[n, b, d] (as packed bf16 pairs)
__global__ void __launch_bounds__(256)
k_permute_query(const float* __restrict__ query, unsigned int* __restrict__ qin) {
    int i = blockIdx.x * 256 + threadIdx.x;        // pair index, 8192*256 total
    const int r = i >> 8, dp = i & 255;
    const float* p = query + ((size_t)(r & 2047) * 4 + (r >> 11)) * 512 + dp * 2;
    qin[(size_t)r * 256 + dp] = pack_bf16(p[0], p[1]);
}

// qT[b][e, n] = q[b*2048+n, e]
__global__ void __launch_bounds__(256)
k_transpose_q(const u16* __restrict__ q, u16* __restrict__ qT) {
    int i = blockIdx.x * 256 + threadIdx.x;        // 4*512*2048 total
    const int b = i >> 20, rem = i & 0xFFFFF;
    const int e = rem >> 11, n = rem & 2047;
    qT[i] = q[((size_t)(b << 11 | n)) * 512 + e];
}

// ------------------------------ GEMM kernels -------------------------------
__global__ void __launch_bounds__(128)
k_linear_in(const u16* __restrict__ qin, const u16* __restrict__ win,
            u16* __restrict__ qbf) {
    __shared__ unsigned int sA[64][20], sB[64][20];
    const int tid = threadIdx.x, lane = tid & 31, warp = tid >> 5;
    const int n0 = blockIdx.x * 64, m0 = blockIdx.y * 64;
    ZERO_ACC(acc);
    gemm_nt_bf16(qin, 512, win, 512, m0, n0, 512, tid, lane, warp, sA, sB, acc);
    const int half = lane >> 4;
    const int row = m0 + warp * 16 + 8 * half, col = n0 + (lane & 15);
#pragma unroll
    for (int t = 0; t < 4; ++t)
#pragma unroll
        for (int v = 0; v < 8; ++v)
            qbf[(size_t)(row + v) * 512 + col + t * 16] = bf16_of(acc[t][v]);
}

__global__ void __launch_bounds__(128)
k_scores(const u16* __restrict__ q, float* __restrict__ scores) {
    __shared__ unsigned int sA[64][20], sB[64][20];
    const int tid = threadIdx.x, lane = tid & 31, warp = tid >> 5;
    const int b = blockIdx.z;
    const int n0 = blockIdx.x * 64, m0 = blockIdx.y * 64;
    const u16* qb = q + (size_t)b * 2048 * 512;
    float* sb = scores + (size_t)b * 2048 * 2048;
    if (n0 >= m0 + 64) {                       // fully masked tile
        for (int idx = tid; idx < 4096; idx += 128)
            sb[(size_t)(m0 + (idx >> 6)) * 2048 + n0 + (idx & 63)] = -1e38f;
        return;
    }
    ZERO_ACC(acc);
    gemm_nt_bf16(qb, 512, qb, 512, m0, n0, 512, tid, lane, warp, sA, sB, acc);
    const int half = lane >> 4;
    const int row = m0 + warp * 16 + 8 * half, col = n0 + (lane & 15);
#pragma unroll
    for (int t = 0; t < 4; ++t)
#pragma unroll
        for (int v = 0; v < 8; ++v) {
            const int gi = row + v, gj = col + t * 16;
            float val = acc[t][v];
            if (gj >= gi) val = -1e38f;
            sb[(size_t)gi * 2048 + gj] = val;
        }
}

__global__ void __launch_bounds__(128)
k_mix(const u16* __restrict__ wgtbf, const u16* __restrict__ qT,
      u16* __restrict__ mixbf) {
    __shared__ unsigned int sA[64][20], sB[64][20];
    const int tid = threadIdx.x, lane = tid & 31, warp = tid >> 5;
    const int b = blockIdx.z;
    const int n0 = blockIdx.x * 64, m0 = blockIdx.y * 64;
    const u16* W = wgtbf + (size_t)b * 2048 * 2048;
    const u16* Bm = qT + (size_t)b * 512 * 2048;
    ZERO_ACC(acc);
    gemm_nt_bf16(W, 2048, Bm, 2048, m0, n0, 2048, tid, lane, warp, sA, sB, acc);
    u16* Mx = mixbf + (size_t)b * 2048 * 512;
    const int half = lane >> 4;
    const int row = m0 + warp * 16 + 8 * half, col = n0 + (lane & 15);
#pragma unroll
    for (int t = 0; t < 4; ++t)
#pragma unroll
        for (int v = 0; v < 8; ++v)
            Mx[(size_t)(row + v) * 512 + col + t * 16] = bf16_of(acc[t][v]);
}

__global__ void __launch_bounds__(128)
k_out(const u16* __restrict__ mixbf, const u16* __restrict__ qbf,
      const u16* __restrict__ wout, float* __restrict__ out) {
    __shared__ unsigned int sA[64][20], sB[64][20];
    const int tid = threadIdx.x, lane = tid & 31, warp = tid >> 5;
    const int n0 = blockIdx.x * 64, m0 = blockIdx.y * 64;
    ZERO_ACC(acc);
    gemm_nt_bf16(mixbf, 512, wout,       1024, m0, n0, 512, tid, lane, warp, sA, sB, acc);
    gemm_nt_bf16(qbf,   512, wout + 512, 1024, m0, n0, 512, tid, lane, warp, sA, sB, acc);
    const float kScale = 1.0507009873554805f, kAlpha = 1.6732632423543772f;
    const int half = lane >> 4;
    const int row = m0 + warp * 16 + 8 * half, col = n0 + (lane & 15);
#pragma unroll
    for (int t = 0; t < 4; ++t)
#pragma unroll
        for (int v = 0; v < 8; ++v) {
            const int r = row + v, e = col + t * 16;
            float x = acc[t][v];
            float y = kScale * (x > 0.f ? x : kAlpha * (__expf(x) - 1.f));
            out[((size_t)(r & 2047) * 4 + (r >> 11)) * 512 + e] = y;
        }
}

// ------------------------------- softmax -----------------------------------
__global__ void __launch_bounds__(256)
k_softmax(float* __restrict__ w, u16* __restrict__ wbf) {
    const int row = blockIdx.x;                 // b*2048 + i
    const int i = row & 2047;
    float* p = w + (size_t)row * 2048;
    u16* pb = wbf + (size_t)row * 2048;
    const int tid = threadIdx.x, lane = tid & 31, wv = tid >> 5;
    if (i == 0) {
        for (int j = tid; j < 2048; j += 256) { p[j] = 0.f; pb[j] = 0; }
        return;
    }
    float vals[8];
    float m = -3.0e38f;
#pragma unroll
    for (int k = 0; k < 8; ++k) { vals[k] = p[tid + k * 256]; m = fmaxf(m, vals[k]); }
#pragma unroll
    for (int off = 16; off > 0; off >>= 1) m = fmaxf(m, __shfl_xor(m, off, 32));
    __shared__ float redm[8];
    if (lane == 0) redm[wv] = m;
    __syncthreads();
    m = redm[0];
#pragma unroll
    for (int k = 1; k < 8; ++k) m = fmaxf(m, redm[k]);
    float s = 0.f;
#pragma unroll
    for (int k = 0; k < 8; ++k) { vals[k] = __expf(vals[k] - m); s += vals[k]; }
#pragma unroll
    for (int off = 16; off > 0; off >>= 1) s += __shfl_xor(s, off, 32);
    __shared__ float reds[8];
    if (lane == 0) reds[wv] = s;
    __syncthreads();
    s = 0.f;
#pragma unroll
    for (int k = 0; k < 8; ++k) s += reds[k];
    const float inv = 1.0f / s;
#pragma unroll
    for (int k = 0; k < 8; ++k) {
        const float r = vals[k] * inv;
        p[tid + k * 256] = r;
        pb[tid + k * 256] = bf16_of(r);
    }
}

// ---------------------------------------------------------------------------
extern "C" void kernel_launch(void* const* d_in, const int* in_sizes, int n_in,
                              void* d_out, int out_size, void* d_ws, size_t ws_size,
                              hipStream_t stream) {
    const float* query = (const float*)d_in[0];   // [2048, 4, 512]
    const float* Win   = (const float*)d_in[1];   // [512, 512]
    const float* Wout  = (const float*)d_in[2];   // [512, 1024]
    float* out     = (float*)d_out;                           // [2048,4,512]
    float* weights = (float*)d_out + (size_t)2048 * 4 * 512;  // [4,2048,2048]

    u16* qin   = (u16*)d_ws;                       // [8192,512]  perm(query) bf16
    u16* win   = qin   + (size_t)8192 * 512;       // [512,512]
    u16* wout  = win   + (size_t)512 * 512;        // [512,1024]
    u16* qbf   = wout  + (size_t)512 * 1024;       // [8192,512]  q bf16
    u16* qT    = qbf   + (size_t)8192 * 512;       // [4][512,2048]
    u16* wgtbf = qT    + (size_t)8192 * 512;       // [4][2048,2048] bf16 weights
    u16* mixbf = wgtbf + (size_t)4 * 2048 * 2048;  // [8192,512]

    k_pack<<<dim3(512),  dim3(256), 0, stream>>>(Win,  (unsigned int*)win,  512 * 512 / 2);
    k_pack<<<dim3(1024), dim3(256), 0, stream>>>(Wout, (unsigned int*)wout, 512 * 1024 / 2);
    k_permute_query<<<dim3(8192), dim3(256), 0, stream>>>(query, (unsigned int*)qin);

    k_linear_in<<<dim3(8, 128),    dim3(128), 0, stream>>>(qin, win, qbf);
    k_transpose_q<<<dim3(16384),   dim3(256), 0, stream>>>(qbf, qT);
    k_scores  <<<dim3(32, 32, 4),  dim3(128), 0, stream>>>(qbf, weights);
    k_softmax <<<dim3(8192),       dim3(256), 0, stream>>>(weights, wgtbf);
    k_mix     <<<dim3(8, 32, 4),   dim3(128), 0, stream>>>(wgtbf, qT, mixbf);
    k_out     <<<dim3(8, 128),     dim3(128), 0, stream>>>(mixbf, qbf, wout, out);
}
